// MultiHeadAttention_37288906063984
// MI455X (gfx1250) — matrix operand order repaired and checked by reading, not compiled, not run on hardware
//
#include <hip/hip_runtime.h>

#define D_MODEL 1024
#define NUM_HEADS 16
#define D_K 64
#define BATCH 4
#define SEQ 2048
#define M_TOT (BATCH * SEQ)   // 8192 rows

typedef _Float16 v8h  __attribute__((ext_vector_type(8)));
typedef _Float16 v16h __attribute__((ext_vector_type(16)));
typedef float    v8f  __attribute__((ext_vector_type(8)));

// ---- WMMA fragment gathers from LDS tiles -------------------------------
// A-matrix 16x32 f16 layout (ISA 7.12.2): lane m = lane&15, half = lane>>4,
// element runs K = [8*half .. +7] and [16+8*half .. +7].
__device__ __forceinline__ v16h a_frag(const _Float16* rowp, int half) {
    v8h lo = *reinterpret_cast<const v8h*>(rowp + 8 * half);
    v8h hi = *reinterpret_cast<const v8h*>(rowp + 16 + 8 * half);
    return __builtin_shufflevector(lo, hi, 0, 1, 2, 3, 4, 5, 6, 7,
                                   8, 9, 10, 11, 12, 13, 14, 15);
}

// B-matrix 32x16 f16 layout: lane n = lane&15, K = [8*half .. +7] and [16+8*half .. +7].
__device__ __forceinline__ v16h b_frag(const _Float16* rowp, int half) {
    v8h lo = *reinterpret_cast<const v8h*>(rowp + 8 * half);
    v8h hi = *reinterpret_cast<const v8h*>(rowp + 16 + 8 * half);
    return __builtin_shufflevector(lo, hi, 0, 1, 2, 3, 4, 5, 6, 7,
                                   8, 9, 10, 11, 12, 13, 14, 15);
}

// ---- DPP16 row reductions (within each 16-lane half of the wave) --------
// xor1 = quad_perm[1,0,3,2]=0xB1; xor2 = quad_perm[2,3,0,1]=0x4E;
// then row_half_mirror (0x141, lane^7) and row_mirror (0x140, lane^15).
// After each stage values are uniform over the combined group, so the
// mirror steps complete the 16-lane reduction. Pure VALU, no DS round-trip.
template <int CTRL>
__device__ __forceinline__ float dpp_bcast_step(float x) {
    union { float f; int i; } u, v;
    u.f = x;
    v.i = __builtin_amdgcn_mov_dpp(u.i, CTRL, 0xF, 0xF, true);
    return v.f;
}
__device__ __forceinline__ float row_max16(float x) {
    x = fmaxf(x, dpp_bcast_step<0xB1>(x));
    x = fmaxf(x, dpp_bcast_step<0x4E>(x));
    x = fmaxf(x, dpp_bcast_step<0x141>(x));
    x = fmaxf(x, dpp_bcast_step<0x140>(x));
    return x;
}
__device__ __forceinline__ float row_sum16(float x) {
    x += dpp_bcast_step<0xB1>(x);
    x += dpp_bcast_step<0x4E>(x);
    x += dpp_bcast_step<0x141>(x);
    x += dpp_bcast_step<0x140>(x);
    return x;
}

// ---- Tiled GEMM: C[M,1024] = A[M,1024] @ W[1024,1024] + bias ------------
// Block: 256 threads = 8 waves; block tile 128x128; wave tile 32x64.
template <bool A_F16, bool OUT_F16>
__global__ __launch_bounds__(256, 1) void gemm_nk1024(
        const void* __restrict__ Ap, const float* __restrict__ W,
        const float* __restrict__ bias, void* __restrict__ Cp) {
    __shared__ alignas(16) _Float16 As[128][32];   // [m][k]
    __shared__ alignas(16) _Float16 Bs[128][32];   // [n][k] (transposed stage)

    const int tid  = threadIdx.x;
    const int lane = tid & 31;
    const int wave = tid >> 5;
    const int half = lane >> 4;
    const int mn   = lane & 15;
    const int m0   = blockIdx.y * 128;
    const int n0   = blockIdx.x * 128;
    const int wm   = (wave >> 1) * 32;   // 0,32,64,96
    const int wn   = (wave & 1) * 64;    // 0,64

    const size_t elt = A_F16 ? sizeof(_Float16) : sizeof(float);
    const char* Abase = (const char*)Ap;

    v8f acc[2][4] = {};

    for (int k0 = 0; k0 < D_MODEL; k0 += 32) {
        __syncthreads();
        for (int i = tid; i < 128 * 32; i += 256) {
            int m = i >> 5, k = i & 31;
            _Float16 v;
            if (A_F16)
                v = ((const _Float16*)Ap)[(size_t)(m0 + m) * D_MODEL + k0 + k];
            else
                v = (_Float16)(((const float*)Ap)[(size_t)(m0 + m) * D_MODEL + k0 + k]);
            As[m][k] = v;
        }
        for (int i = tid; i < 32 * 128; i += 256) {
            int k = i >> 7, n = i & 127;
            Bs[n][k] = (_Float16)W[(size_t)(k0 + k) * D_MODEL + n0 + n];
        }
        if (k0 + 32 < D_MODEL) {   // prefetch next k-tile (global_prefetch_b8)
            int m  = tid >> 1, kq = (tid & 1) * 16;
            __builtin_prefetch(Abase + ((size_t)(m0 + m) * D_MODEL + k0 + 32 + kq) * elt, 0, 1);
            int kb = tid >> 3, nb = (tid & 7) * 16;
            __builtin_prefetch(&W[(size_t)(k0 + 32 + kb) * D_MODEL + n0 + nb], 0, 1);
        }
        __syncthreads();

        v16h a[2], b[4];
        for (int tm = 0; tm < 2; ++tm)
            a[tm] = a_frag(&As[wm + tm * 16 + mn][0], half);
        for (int tn = 0; tn < 4; ++tn)
            b[tn] = b_frag(&Bs[wn + tn * 16 + mn][0], half);

        for (int tm = 0; tm < 2; ++tm)
            for (int tn = 0; tn < 4; ++tn)
                acc[tm][tn] = __builtin_amdgcn_wmma_f32_16x16x32_f16(
                    false, a[tm], false, b[tn], (short)0, acc[tm][tn], false, false);
    }

    for (int tm = 0; tm < 2; ++tm)
        for (int tn = 0; tn < 4; ++tn) {
            int col  = n0 + wn + tn * 16 + mn;
            float bv = bias[col];
            for (int r = 0; r < 8; ++r) {
                int row = m0 + wm + tm * 16 + r + 8 * half;
                float v = acc[tm][tn][r] + bv;
                if (OUT_F16)
                    ((_Float16*)Cp)[(size_t)row * D_MODEL + col] = (_Float16)v;
                else
                    ((float*)Cp)[(size_t)row * D_MODEL + col] = v;
            }
        }
}

// ---- Flash attention: per (b,h), 128 query rows per block ---------------
__global__ __launch_bounds__(256, 1) void attention_kernel(
        const _Float16* __restrict__ qf, const _Float16* __restrict__ kf,
        const _Float16* __restrict__ vf, _Float16* __restrict__ of) {
    __shared__ alignas(16) _Float16 Qs[128][64];     // [m][d], pre-scaled by 1/8
    __shared__ alignas(16) _Float16 Ks[64][64];      // [n(key)][d]  (B-side of Q@K^T)
    __shared__ alignas(16) _Float16 VT[64][64];      // [d][s(key)]  (B-side of P@V)
    __shared__ alignas(16) _Float16 Ps[8][16][64];   // per-wave P bounce (C->A layout)

    const int tid  = threadIdx.x;
    const int lane = tid & 31;
    const int wave = tid >> 5;
    const int half = lane >> 4;
    const int mn   = lane & 15;
    const int bh   = blockIdx.y;
    const int b    = bh >> 4;
    const int h    = bh & 15;
    const int q0   = blockIdx.x * 128;
    const size_t rowbase = (size_t)b * SEQ;
    const int hc   = h * D_K;

    // Stage Q once, folding the 1/sqrt(d_k)=0.125 scale (exact in f16).
    for (int i = tid; i < 128 * 64; i += 256) {
        int m = i >> 6, d = i & 63;
        Qs[m][d] = qf[(rowbase + q0 + m) * D_MODEL + hc + d] * (_Float16)0.125f;
    }
    __syncthreads();

    const int wm = wave * 16;
    v16h aq[2];
    aq[0] = a_frag(&Qs[wm + mn][0], half);
    aq[1] = a_frag(&Qs[wm + mn][32], half);

    float mi[8], li[8];
    v8f o[4] = {};
    for (int r = 0; r < 8; ++r) { mi[r] = -1e30f; li[r] = 0.0f; }

    const unsigned long long kbase = (unsigned long long)(uintptr_t)kf;

    for (int s0 = 0; s0 < SEQ; s0 += 64) {
        __syncthreads();   // previous iteration done with Ks/VT

        // --- K tile: async DMA global->LDS, 16B per lane-transfer.
        // GVS mode: vdst = LDS byte address, vaddr = 32-bit offset, saddr = base.
        for (int j = 0; j < 2; ++j) {
            int idx = tid * 2 + j;              // 0..511
            int n   = idx >> 3;                  // key row 0..63
            int d0  = (idx & 7) * 8;             // 8 halves = 16 bytes
            unsigned lds  = (unsigned)(uintptr_t)&Ks[n][d0];
            unsigned voff = (unsigned)(((rowbase + s0 + n) * D_MODEL + hc + d0)
                                       * sizeof(_Float16));
            asm volatile("global_load_async_to_lds_b128 %0, %1, %2"
                         :: "v"(lds), "v"(voff), "s"(kbase) : "memory");
        }
        // --- V tile: manual transposed stage (DMA cannot transpose).
        for (int i = tid; i < 64 * 64; i += 256) {
            int n = i >> 6, d = i & 63;
            VT[d][n] = vf[(rowbase + s0 + n) * D_MODEL + hc + d];
        }
        if (s0 + 64 < SEQ) {   // prefetch next key chunk
            int t = tid & 127, n = t >> 1, off = (t & 1) * 32;
            const _Float16* src = (tid < 128) ? kf : vf;
            __builtin_prefetch(&src[(rowbase + s0 + 64 + n) * D_MODEL + hc + off], 0, 1);
        }
        asm volatile("s_wait_asynccnt 0x0" ::: "memory");
        __syncthreads();

        // Scores: 16x64 tile = 4 n-tiles, K=64 (2 wmma each), f32 accum.
        v8f cs[4] = {};
        for (int nt = 0; nt < 4; ++nt)
            for (int kk = 0; kk < 2; ++kk) {
                v16h bk = b_frag(&Ks[nt * 16 + mn][kk * 32], half);
                cs[nt] = __builtin_amdgcn_wmma_f32_16x16x32_f16(
                    false, aq[kk], false, bk, (short)0, cs[nt], false, false);
            }

        // Online softmax; row r+8*half stats uniform across its 16 lanes.
        float mnew[8], corr[8], rs[8];
        for (int r = 0; r < 8; ++r) {
            float rm = fmaxf(fmaxf(cs[0][r], cs[1][r]), fmaxf(cs[2][r], cs[3][r]));
            rm      = row_max16(rm);
            mnew[r] = fmaxf(mi[r], rm);
            corr[r] = __expf(mi[r] - mnew[r]);
            rs[r]   = 0.0f;
        }
        for (int nt = 0; nt < 4; ++nt)
            for (int r = 0; r < 8; ++r) {
                float p = __expf(cs[nt][r] - mnew[r]);
                cs[nt][r] = p;
                rs[r] += p;
            }
        for (int r = 0; r < 8; ++r) {
            rs[r] = row_sum16(rs[r]);
            li[r] = li[r] * corr[r] + rs[r];
            mi[r] = mnew[r];
            for (int dt = 0; dt < 4; ++dt) o[dt][r] *= corr[r];
        }

        // Bounce P through per-wave LDS region (C-layout -> A-layout).
        // Wave-private: same-wave DS ops complete in order, so no workgroup
        // barrier is needed; wave_barrier() stops compiler reordering.
        for (int nt = 0; nt < 4; ++nt)
            for (int r = 0; r < 8; ++r)
                Ps[wave][r + 8 * half][nt * 16 + mn] = (_Float16)cs[nt][r];
        __builtin_amdgcn_wave_barrier();

        v16h ap[2];
        ap[0] = a_frag(&Ps[wave][mn][0], half);
        ap[1] = a_frag(&Ps[wave][mn][32], half);
        for (int dt = 0; dt < 4; ++dt)
            for (int kk = 0; kk < 2; ++kk) {
                v16h bv = b_frag(&VT[dt * 16 + mn][kk * 32], half);
                o[dt] = __builtin_amdgcn_wmma_f32_16x16x32_f16(
                    false, ap[kk], false, bv, (short)0, o[dt], false, false);
            }
    }

    // Normalize and store f16 attention output (concat-head layout).
    for (int dt = 0; dt < 4; ++dt)
        for (int r = 0; r < 8; ++r) {
            int row = q0 + wm + r + 8 * half;
            float v = o[dt][r] / li[r];
            of[(rowbase + row) * D_MODEL + hc + dt * 16 + mn] = (_Float16)v;
        }
}

// ---- Host launcher -------------------------------------------------------
extern "C" void kernel_launch(void* const* d_in, const int* in_sizes, int n_in,
                              void* d_out, int out_size, void* d_ws, size_t ws_size,
                              hipStream_t stream) {
    const float* Q  = (const float*)d_in[0];
    const float* K  = (const float*)d_in[1];
    const float* V  = (const float*)d_in[2];
    const float* Wq = (const float*)d_in[3];
    const float* bq = (const float*)d_in[4];
    const float* Wk = (const float*)d_in[5];
    const float* bk = (const float*)d_in[6];
    const float* Wv = (const float*)d_in[7];
    const float* bv = (const float*)d_in[8];
    const float* Wo = (const float*)d_in[9];
    const float* bo = (const float*)d_in[10];
    float* out = (float*)d_out;

    char* ws = (char*)d_ws;
    const size_t tbytes = (size_t)M_TOT * D_MODEL * sizeof(_Float16);  // 16 MB
    _Float16* qf = (_Float16*)(ws);
    _Float16* kf = (_Float16*)(ws + tbytes);
    _Float16* vf = (_Float16*)(ws + 2 * tbytes);
    _Float16* af = (_Float16*)(ws + 3 * tbytes);

    dim3 block(256);
    dim3 ggrid(D_MODEL / 128, M_TOT / 128);
    gemm_nk1024<false, true><<<ggrid, block, 0, stream>>>(Q, Wq, bq, qf);
    gemm_nk1024<false, true><<<ggrid, block, 0, stream>>>(K, Wk, bk, kf);
    gemm_nk1024<false, true><<<ggrid, block, 0, stream>>>(V, Wv, bv, vf);

    dim3 agrid(SEQ / 128, BATCH * NUM_HEADS);
    attention_kernel<<<agrid, block, 0, stream>>>(qf, kf, vf, af);

    gemm_nk1024<true, false><<<ggrid, block, 0, stream>>>(af, Wo, bo, out);
}
